// NonLocalBlock_53317724012983
// MI455X (gfx1250) — matrix operand order repaired and checked by reading, not compiled, not run on hardware
//
#include <hip/hip_runtime.h>
#include <hip/hip_bf16.h>
#include <math.h>

// Problem sizes (fixed by the reference)
#define BB 16
#define CC 256
#define NN 2304            // 48*48
#define NT 144             // NN/16
#define JBLKS 72           // NN/32

typedef __attribute__((ext_vector_type(16))) __bf16 v16bf;
typedef __attribute__((ext_vector_type(8)))  __bf16 v8bf;
typedef __attribute__((ext_vector_type(8)))  float  v8f;
typedef __attribute__((ext_vector_type(4)))  float  v4f;

static __device__ inline v16bf cat8(v8bf lo, v8bf hi) {
    return __builtin_shufflevector(lo, hi, 0,1,2,3,4,5,6,7,8,9,10,11,12,13,14,15);
}

// Async DMA: global memory -> LDS, 16 bytes, tracked by ASYNCcnt (in-order).
// ldsOff = LDS byte offset (low 32 bits of a generic pointer into __shared__),
// gptr   = global address. No VGPR round-trip.
static __device__ inline void async_g2l_b128(unsigned ldsOff, const void* gptr) {
    asm volatile("global_load_async_to_lds_b128 %0, %1, off"
                 :: "v"(ldsOff), "v"((unsigned long long)(uintptr_t)gptr)
                 : "memory");
}
static __device__ inline void wait_async0() {   // all async DMAs landed
    asm volatile("s_wait_asynccnt 0x0" ::: "memory");
}
static __device__ inline void wait_async8() {   // all but the newest 8 landed
    asm volatile("s_wait_asynccnt 0x8" ::: "memory");
}

// ---------------------------------------------------------------------------
// Projection kernel:  theta/phi/g = W (CxC) * x (CxN)  per batch, bf16 outputs
//   Q  [b][n][c]  = theta^T   (row n contiguous in c)
//   Km [b][n][c]  = phi^T     (row n contiguous in c)
//   G  [b][c][n]  = g         (row c contiguous in n)
// One wave handles one 16-wide n-tile of one batch; 4 waves / block.
// WMMA: D(16o x 16n) = A(W tile 16o x 32c) * B(x tile 32c x 16n), 8 K-steps.
// ---------------------------------------------------------------------------
__global__ __launch_bounds__(128)
void nlb_proj_kernel(const float* __restrict__ x,
                     const float* __restrict__ w_theta,
                     const float* __restrict__ w_phi,
                     const float* __restrict__ w_g,
                     __bf16* __restrict__ Q,
                     __bf16* __restrict__ Km,
                     __bf16* __restrict__ G)
{
    const int lane = threadIdx.x & 31;
    const int wv   = threadIdx.x >> 5;
    const int ntg  = blockIdx.x * 4 + wv;       // global n-tile id: [0, BB*NT)
    const int b    = ntg / NT;
    const int n0   = (ntg % NT) * 16;

    const int nIdx = n0 + (lane & 15);          // this lane's column n
    const int cgrp = (lane >> 4) * 16;          // B-operand K sub-range

    // Preload x B-operands for all 8 K-chunks: B[k=c][n], lane holds column n,
    // 16 contiguous c values (strided gathers from x[c][n], f32 -> bf16).
    const float* xb = x + (size_t)b * CC * NN;
    v16bf xB[8];
    #pragma unroll
    for (int kc = 0; kc < 8; ++kc) {
        const int cbase = kc * 32 + cgrp;
        #pragma unroll
        for (int e = 0; e < 16; ++e)
            xB[kc][e] = (__bf16)xb[(size_t)(cbase + e) * NN + nIdx];
    }

    const int rbase = (lane >> 4) * 16;         // A-operand K sub-range

    // widx fully unrolled so each weight pointer stays a kernarg-derived
    // global pointer (keeps loads as global_load_*, not flat_load_*).
    #pragma unroll
    for (int widx = 0; widx < 3; ++widx) {
        const float* wm = (widx == 0) ? w_theta : (widx == 1) ? w_phi : w_g;
        for (int ot = 0; ot < 16; ++ot) {
            v8f acc = {};
            #pragma unroll
            for (int kc = 0; kc < 8; ++kc) {
                // A tile: row o = ot*16 + (lane&15); K = kc*32 + rbase + {0..15}
                const float* ar = wm + (size_t)(ot * 16 + (lane & 15)) * CC
                                     + kc * 32 + rbase;
                v16bf a;
                #pragma unroll
                for (int e = 0; e < 8; ++e) a[e]     = (__bf16)ar[e];
                #pragma unroll
                for (int e = 0; e < 8; ++e) a[8 + e] = (__bf16)ar[8 + e];
                acc = __builtin_amdgcn_wmma_f32_16x16x32_bf16(
                          false, a, false, xB[kc], (short)0, acc, false, false);
            }
            // D layout: lane -> n = nIdx ; VGPR r -> o = ot*16 + 8*(lane>>4) + r
            const int oBase = ot * 16 + 8 * (lane >> 4);
            if (widx == 0) {
                v8bf o8;
                #pragma unroll
                for (int r = 0; r < 8; ++r) o8[r] = (__bf16)acc[r];
                *(v8bf*)(Q + ((size_t)b * NN + nIdx) * CC + oBase) = o8;
            } else if (widx == 1) {
                v8bf o8;
                #pragma unroll
                for (int r = 0; r < 8; ++r) o8[r] = (__bf16)acc[r];
                *(v8bf*)(Km + ((size_t)b * NN + nIdx) * CC + oBase) = o8;
            } else {
                #pragma unroll
                for (int r = 0; r < 8; ++r)
                    G[((size_t)b * CC + oBase + r) * NN + nIdx] = (__bf16)acc[r];
            }
        }
    }
}

// ---------------------------------------------------------------------------
// Fused flash-attention kernel with double-buffered async-LDS staging.
//  8 waves/block = 4 i-tiles x 2 c-halves; block covers 64 rows of one batch.
//  Per 32-wide j-block: K-slice [32j][256c] and g-slice [256c][32j] live in
//  LDS; the next block's slices are DMA'd (ASYNCcnt, in-order completion)
//  while the current block computes:
//   GEMM1: S^T(16j x 16i) = Krows(A) x Q^T(B), 8 K-steps, two j-subtiles.
//   Online softmax per lane (lane&15 == i), combine halves via shfl_xor(16).
//   P packs directly into the A operand of GEMM2 (no cross-lane movement).
//   GEMM2: O(16i x 16c) += P(A) x G^T(B) for this wave's 8 c-tiles.
// ---------------------------------------------------------------------------
__global__ __launch_bounds__(256)
void nlb_attn_kernel(const __bf16* __restrict__ Q,
                     const __bf16* __restrict__ Km,
                     const __bf16* __restrict__ G,
                     float* __restrict__ out)
{
    __shared__ __bf16 KsAll[2][32 * 256];   // [buf][j][c]  2 x 16 KB
    __shared__ __bf16 GsAll[2][256 * 32];   // [buf][c][j]  2 x 16 KB

    const int lane = threadIdx.x & 31;
    const int wv   = threadIdx.x >> 5;
    const int iw   = wv & 3;          // i sub-tile
    const int ch   = wv >> 2;         // c half (0 -> c 0..127, 1 -> 128..255)

    const int b  = blockIdx.x / 36;
    const int i0 = (blockIdx.x % 36) * 64 + iw * 16;

    const int l15  = lane & 15;
    const int lhi  = lane >> 4;
    const int rb8  = lhi * 8;

    // Preload Q as B-operand: lane holds column i, K = c in {8h..8h+7, 16+8h..16+8h+7}.
    const __bf16* qrow = Q + ((size_t)b * NN + i0 + l15) * CC + lhi * 8;
    v16bf qB[8];
    #pragma unroll
    for (int kc = 0; kc < 8; ++kc) {
        v8bf lo = *(const v8bf*)(qrow + kc * 32);
        v8bf hi = *(const v8bf*)(qrow + kc * 32 + 16);
        qB[kc] = cat8(lo, hi);
    }

    v8f Oacc[8];
    #pragma unroll
    for (int ct = 0; ct < 8; ++ct) Oacc[ct] = (v8f){};

    float m    = -__builtin_inff();
    float lsum = 0.0f;

    const __bf16* kbase = Km + (size_t)b * NN * CC;
    const __bf16* gbase = G  + (size_t)b * CC * NN;

    // per-thread staging coordinates (8 x b128 per thread per j-block)
    const int tid  = threadIdx.x;

    // issue async DMA of one j-block's K/g slices into buffer `buf`
    auto stage_issue = [&](int j0, int buf) {
        const __bf16* ks = kbase + (size_t)j0 * CC;     // [32 rows][256 c]
        __bf16* kd = KsAll[buf];
        #pragma unroll
        for (int it = 0; it < 4; ++it) {
            int idx = tid + it * 256;                   // 0..1023 16B-chunks
            int row = idx >> 5, grp = idx & 31;
            async_g2l_b128((unsigned)(uintptr_t)(kd + row * 256 + grp * 8),
                           ks + (size_t)row * CC + grp * 8);
        }
        const __bf16* gs = gbase + j0;                  // [256 c][32 j]
        __bf16* gd = GsAll[buf];
        #pragma unroll
        for (int it = 0; it < 4; ++it) {
            int idx = tid + it * 256;
            int c = idx >> 2, grp = idx & 3;
            async_g2l_b128((unsigned)(uintptr_t)(gd + c * 32 + grp * 8),
                           gs + (size_t)c * NN + grp * 8);
        }
    };

    stage_issue(0, 0);                                  // prologue: jb = 0

    for (int jb = 0; jb < JBLKS; ++jb) {
        const int j0 = jb * 32;

        // Overlap: kick off next block's DMA into the other buffer.
        // (Bottom barrier of the previous iteration guarantees nobody still
        //  reads that buffer.)  In-order ASYNCcnt completion => waiting for
        //  "<= 8 outstanding" means the current buffer's 8 loads are done.
        if (jb + 1 < JBLKS) {
            stage_issue(j0 + 32, (jb + 1) & 1);
            wait_async8();
        } else {
            wait_async0();
        }
        if (jb + 2 < JBLKS) {   // L2 hint one block ahead of the DMA
            __builtin_prefetch(kbase + (size_t)(j0 + 64) * CC + lane * 64, 0, 0);
            __builtin_prefetch(gbase + (size_t)lane * 8 * NN + j0 + 64, 0, 0);
        }
        __syncthreads();        // everyone's current-buffer data visible

        const __bf16* Ks = KsAll[jb & 1];
        const __bf16* Gs = GsAll[jb & 1];

        // ---- GEMM1: two S^T subtiles ----
        v8f st0 = (v8f){}, st1 = (v8f){};
        #pragma unroll
        for (int kc = 0; kc < 8; ++kc) {
            const __bf16* a0 = Ks + l15 * 256 + kc * 32 + rb8;
            v16bf A0 = cat8(*(const v8bf*)a0, *(const v8bf*)(a0 + 16));
            st0 = __builtin_amdgcn_wmma_f32_16x16x32_bf16(
                      false, A0, false, qB[kc], (short)0, st0, false, false);
            const __bf16* a1 = a0 + 16 * 256;
            v16bf A1 = cat8(*(const v8bf*)a1, *(const v8bf*)(a1 + 16));
            st1 = __builtin_amdgcn_wmma_f32_16x16x32_bf16(
                      false, A1, false, qB[kc], (short)0, st1, false, false);
        }

        // ---- online softmax (per lane: row i = lane&15) ----
        float tmax = -__builtin_inff();
        #pragma unroll
        for (int r = 0; r < 8; ++r)
            tmax = fmaxf(tmax, fmaxf(st0[r], st1[r]));
        tmax = fmaxf(tmax, __shfl_xor(tmax, 16, 32));
        const float mnew  = fmaxf(m, tmax);
        const float alpha = __expf(m - mnew);

        float p0[8], p1[8], ps = 0.0f;
        #pragma unroll
        for (int r = 0; r < 8; ++r) {
            p0[r] = __expf(st0[r] - mnew);
            p1[r] = __expf(st1[r] - mnew);
            ps += p0[r] + p1[r];
        }
        ps += __shfl_xor(ps, 16, 32);
        lsum = lsum * alpha + ps;
        m = mnew;

        // redistribute alpha from (i = lane&15) domain to O-accum rows
        float ar[8];
        #pragma unroll
        for (int r = 0; r < 8; ++r) ar[r] = __shfl(alpha, rb8 + r, 32);
        #pragma unroll
        for (int ct = 0; ct < 8; ++ct)
            #pragma unroll
            for (int r = 0; r < 8; ++r) Oacc[ct][r] *= ar[r];

        // P already matches GEMM2's A-operand layout: pack to bf16
        v16bf pA;
        #pragma unroll
        for (int e = 0; e < 8; ++e) pA[e]     = (__bf16)p0[e];
        #pragma unroll
        for (int e = 0; e < 8; ++e) pA[8 + e] = (__bf16)p1[e];

        // ---- GEMM2: O += P x G^T over this wave's 8 c-tiles ----
        #pragma unroll
        for (int ct = 0; ct < 8; ++ct) {
            const int c = ch * 128 + ct * 16 + l15;
            const __bf16* gp = Gs + c * 32 + lhi * 8;
            v16bf Bg = cat8(*(const v8bf*)gp, *(const v8bf*)(gp + 16));
            Oacc[ct] = __builtin_amdgcn_wmma_f32_16x16x32_bf16(
                           false, pA, false, Bg, (short)0, Oacc[ct], false, false);
        }
        __syncthreads();        // LDS reads done before next overwrite
    }

    // ---- epilogue: divide by softmax denominator, store f32 output ----
    float linv[8];
    #pragma unroll
    for (int r = 0; r < 8; ++r)
        linv[r] = 1.0f / __shfl(lsum, rb8 + r, 32);

    const int iBase = i0 + 8 * lhi;
    #pragma unroll
    for (int ct = 0; ct < 8; ++ct) {
        const int c = ch * 128 + ct * 16 + l15;
        float* dst = out + ((size_t)b * CC + c) * NN + iBase;
        v4f lo, hi;
        #pragma unroll
        for (int r = 0; r < 4; ++r) {
            lo[r] = Oacc[ct][r]     * linv[r];
            hi[r] = Oacc[ct][4 + r] * linv[4 + r];
        }
        *(v4f*)dst       = lo;
        *(v4f*)(dst + 4) = hi;
    }
}

// ---------------------------------------------------------------------------
extern "C" void kernel_launch(void* const* d_in, const int* in_sizes, int n_in,
                              void* d_out, int out_size, void* d_ws, size_t ws_size,
                              hipStream_t stream) {
    const float* x  = (const float*)d_in[0];
    const float* wt = (const float*)d_in[1];
    const float* wp = (const float*)d_in[2];
    const float* wg = (const float*)d_in[3];
    float* out = (float*)d_out;

    // Workspace: three bf16 tensors of B*N*C (Q = theta^T, K = phi^T, G = g)
    const size_t elems = (size_t)BB * NN * CC;   // 9,437,184
    __bf16* Q  = (__bf16*)d_ws;
    __bf16* Km = Q + elems;
    __bf16* G  = Km + elems;                      // total 3*elems*2 = ~56.6 MB

    // 1) projections: BB*NT n-tiles, 4 waves/block
    nlb_proj_kernel<<<(BB * NT) / 4, 128, 0, stream>>>(x, wt, wp, wg, Q, Km, G);

    // 2) fused attention: 36 blocks of 64 rows per batch
    nlb_attn_kernel<<<BB * 36, 256, 0, stream>>>(Q, Km, G, out);
}
